// SAGE_1288490189413
// MI455X (gfx1250) — matrix operand order repaired and checked by reading, not compiled, not run on hardware
//
#include <hip/hip_runtime.h>

typedef __attribute__((ext_vector_type(16))) __bf16 v16bf;
typedef __attribute__((ext_vector_type(8)))  float  v8f;

#define FOUT 256

// ---------------------------------------------------------------------------
// Degree count (integer atomics into the inv_deg buffer, reinterpreted)
// ---------------------------------------------------------------------------
__global__ void degree_kernel(const int* __restrict__ edst,
                              unsigned* __restrict__ deg, int n_edges) {
  int i = blockIdx.x * blockDim.x + threadIdx.x;
  int stride = gridDim.x * blockDim.x;
  for (; i < n_edges; i += stride)
    atomicAdd(&deg[edst[i]], 1u);
}

__global__ void inv_deg_kernel(float* __restrict__ buf, int n) {
  int i = blockIdx.x * blockDim.x + threadIdx.x;
  if (i < n) {
    unsigned u = ((const unsigned*)buf)[i];
    buf[i] = 1.0f / fmaxf((float)u, 1.0f);
  }
}

// ---------------------------------------------------------------------------
// Weight prep: transpose W[K][N] -> WT[N][K] and split fp32 into bf16 hi/lo
// ---------------------------------------------------------------------------
__global__ void wt_split_kernel(const float* __restrict__ W,
                                __bf16* __restrict__ hi, __bf16* __restrict__ lo,
                                int K, int N) {
  int i = blockIdx.x * blockDim.x + threadIdx.x;
  if (i < K * N) {
    int k = i / N, n = i - k * N;
    float w = W[i];
    __bf16 h = (__bf16)w;
    float r = w - (float)h;
    hi[(size_t)n * K + k] = h;
    lo[(size_t)n * K + k] = (__bf16)r;
  }
}

// ---------------------------------------------------------------------------
// Edge scatter: S[dst] += X[src], one wave per edge, fp32 atomics (L2-resident)
// ---------------------------------------------------------------------------
template<int FIN>
__global__ void scatter_kernel(const float* __restrict__ X,
                               const int* __restrict__ esrc,
                               const int* __restrict__ edst,
                               float* __restrict__ S, int n_edges) {
  int wid  = (blockIdx.x * blockDim.x + threadIdx.x) >> 5;
  int lane = threadIdx.x & 31;
  int nw   = (gridDim.x * blockDim.x) >> 5;
  for (int e = wid; e < n_edges; e += nw) {
    int s = esrc[e], d = edst[e];
    const float4* xs = (const float4*)(X + (size_t)s * FIN);
    float* sd = S + (size_t)d * FIN;
#pragma unroll
    for (int f = 0; f < FIN / 128; ++f) {
      int idx = f * 32 + lane;
      float4 v = xs[idx];
      unsafeAtomicAdd(sd + idx * 4 + 0, v.x);
      unsafeAtomicAdd(sd + idx * 4 + 1, v.y);
      unsafeAtomicAdd(sd + idx * 4 + 2, v.z);
      unsafeAtomicAdd(sd + idx * 4 + 3, v.w);
    }
  }
}

// ---------------------------------------------------------------------------
// Fused SAGE GEMM: Y = relu?( (S .* inv_deg) @ Wl + X @ Wr + b )
// bf16x3 emulation of fp32 on v_wmma_f32_16x16x32_bf16.
// Block = 4 waves; wave -> 16 rows x 64 cols; block -> 16 rows x 256 cols.
// ---------------------------------------------------------------------------
template<int FIN, bool RELU>
__global__ __launch_bounds__(128)
void sage_gemm_kernel(const float* __restrict__ S, const float* __restrict__ X,
                      const float* __restrict__ invdeg,
                      const __bf16* __restrict__ WlT_hi, const __bf16* __restrict__ WlT_lo,
                      const __bf16* __restrict__ WrT_hi, const __bf16* __restrict__ WrT_lo,
                      const float* __restrict__ bias, float* __restrict__ Y, int n) {
  const int wave = threadIdx.x >> 5;
  const int lane = threadIdx.x & 31;
  const int m    = lane & 15;     // row-in-tile (A), col-in-tile (B/C)
  const int half = lane >> 4;     // K-half selector per ISA fragment layout
  const int r0   = blockIdx.x * 16;
  if (r0 >= n) return;
  const int col0 = wave * 64;
  const int row  = min(r0 + m, n - 1);
  const float idg = invdeg[row];

  v8f acc[4] = {};

#pragma unroll
  for (int p = 0; p < 2; ++p) {
    const float*  __restrict__ A  = p ? X : S;
    const __bf16* __restrict__ BH = p ? WrT_hi : WlT_hi;
    const __bf16* __restrict__ BL = p ? WrT_lo : WlT_lo;
    const float scale = p ? 1.0f : idg;
    const float* arow = A + (size_t)row * FIN;

    for (int k0 = 0; k0 < FIN; k0 += 32) {
      // A fragment (16x32 bf16 layout): lane holds K = half*16 + 0..15
      const float* b0 = arow + k0 + half * 16;
      float4 q0 = *(const float4*)(b0);
      float4 q1 = *(const float4*)(b0 + 4);
      float4 q2 = *(const float4*)(b0 + 8);
      float4 q3 = *(const float4*)(b0 + 12);
      float a[16] = {q0.x, q0.y, q0.z, q0.w, q1.x, q1.y, q1.z, q1.w,
                     q2.x, q2.y, q2.z, q2.w, q3.x, q3.y, q3.z, q3.w};
      v16bf ahi, alo;
#pragma unroll
      for (int i = 0; i < 16; ++i) {
        float v = a[i] * scale;
        __bf16 h = (__bf16)v;
        ahi[i] = h;
        alo[i] = (__bf16)(v - (float)h);
      }
#pragma unroll
      for (int t = 0; t < 4; ++t) {
        const int col = col0 + t * 16 + m;
        // B fragment (32x16): lane holds col, K = half*16 + 0..15 (contiguous in WT)
        const __bf16* pb = BH + (size_t)col * FIN + k0 + half * 16;
        const __bf16* pl = BL + (size_t)col * FIN + k0 + half * 16;
        v16bf bhi = *(const v16bf*)pb;
        v16bf blo = *(const v16bf*)pl;
        acc[t] = __builtin_amdgcn_wmma_f32_16x16x32_bf16(false, ahi, false, bhi,
                                                         (short)0, acc[t], false, false);
        acc[t] = __builtin_amdgcn_wmma_f32_16x16x32_bf16(false, alo, false, bhi,
                                                         (short)0, acc[t], false, false);
        acc[t] = __builtin_amdgcn_wmma_f32_16x16x32_bf16(false, ahi, false, blo,
                                                         (short)0, acc[t], false, false);
      }
    }
  }

  // Epilogue: C layout -> VGPR i: lanes0-15 row=r0+i, lanes16-31 row=r0+8+i
#pragma unroll
  for (int t = 0; t < 4; ++t) {
    const int col = col0 + t * 16 + m;
    const float b = bias[col];
#pragma unroll
    for (int i = 0; i < 8; ++i) {
      int rr = r0 + i + half * 8;
      float v = acc[t][i] + b;
      if (RELU) v = fmaxf(v, 0.0f);
      if (rr < n) Y[(size_t)rr * FOUT + col] = v;
    }
  }
}

// ---------------------------------------------------------------------------
extern "C" void kernel_launch(void* const* d_in, const int* in_sizes, int n_in,
                              void* d_out, int out_size, void* d_ws, size_t ws_size,
                              hipStream_t stream) {
  const float* x   = (const float*)d_in[0];
  const float* Wl0 = (const float*)d_in[1];
  const float* bl0 = (const float*)d_in[2];
  const float* Wr0 = (const float*)d_in[3];
  const float* Wl1 = (const float*)d_in[4];
  const float* bl1 = (const float*)d_in[5];
  const float* Wr1 = (const float*)d_in[6];
  const float* Wl2 = (const float*)d_in[7];
  const float* bl2 = (const float*)d_in[8];
  const float* Wr2 = (const float*)d_in[9];
  const int* esrc  = (const int*)d_in[10];
  const int* edst  = (const int*)d_in[11];

  const int IN_C = 128, HID = 256;
  const int n  = in_sizes[0] / IN_C;
  const int ne = in_sizes[10];

  // ---- workspace carve-out ----
  char* w = (char*)d_ws;
  auto carve = [&](size_t bytes) -> char* {
    char* p = w;
    w += (bytes + 255) & ~(size_t)255;
    return p;
  };
  float* S   = (float*)carve((size_t)n * HID * 4);   // scatter accumulator (max FIN)
  float* H1  = (float*)carve((size_t)n * HID * 4);   // layer-0 output
  float* H2  = (float*)carve((size_t)n * HID * 4);   // layer-1 output
  float* idg = (float*)carve((size_t)n * 4);         // deg -> inv_deg (in place)

  __bf16* wl0h = (__bf16*)carve((size_t)IN_C * HID * 2);
  __bf16* wl0l = (__bf16*)carve((size_t)IN_C * HID * 2);
  __bf16* wr0h = (__bf16*)carve((size_t)IN_C * HID * 2);
  __bf16* wr0l = (__bf16*)carve((size_t)IN_C * HID * 2);
  __bf16* wl1h = (__bf16*)carve((size_t)HID * HID * 2);
  __bf16* wl1l = (__bf16*)carve((size_t)HID * HID * 2);
  __bf16* wr1h = (__bf16*)carve((size_t)HID * HID * 2);
  __bf16* wr1l = (__bf16*)carve((size_t)HID * HID * 2);
  __bf16* wl2h = (__bf16*)carve((size_t)HID * HID * 2);
  __bf16* wl2l = (__bf16*)carve((size_t)HID * HID * 2);
  __bf16* wr2h = (__bf16*)carve((size_t)HID * HID * 2);
  __bf16* wr2l = (__bf16*)carve((size_t)HID * HID * 2);

  // ---- one-time per call: degrees + weight transpose/split ----
  hipMemsetAsync(idg, 0, (size_t)n * 4, stream);
  degree_kernel<<<1024, 256, 0, stream>>>(edst, (unsigned*)idg, ne);
  inv_deg_kernel<<<(n + 255) / 256, 256, 0, stream>>>(idg, n);

  auto split = [&](const float* W, __bf16* h, __bf16* l, int K, int N) {
    int tot = K * N;
    wt_split_kernel<<<(tot + 255) / 256, 256, 0, stream>>>(W, h, l, K, N);
  };
  split(Wl0, wl0h, wl0l, IN_C, HID);
  split(Wr0, wr0h, wr0l, IN_C, HID);
  split(Wl1, wl1h, wl1l, HID, HID);
  split(Wr1, wr1h, wr1l, HID, HID);
  split(Wl2, wl2h, wl2l, HID, HID);
  split(Wr2, wr2h, wr2l, HID, HID);

  const int gemm_grid = (n + 15) / 16;

  // ---- layer 0: x[100k,128] -> H1[100k,256], relu ----
  hipMemsetAsync(S, 0, (size_t)n * IN_C * 4, stream);
  scatter_kernel<128><<<4096, 256, 0, stream>>>(x, esrc, edst, S, ne);
  sage_gemm_kernel<128, true><<<gemm_grid, 128, 0, stream>>>(
      S, x, idg, wl0h, wl0l, wr0h, wr0l, bl0, H1, n);

  // ---- layer 1: H1 -> H2, relu ----
  hipMemsetAsync(S, 0, (size_t)n * HID * 4, stream);
  scatter_kernel<256><<<4096, 256, 0, stream>>>(H1, esrc, edst, S, ne);
  sage_gemm_kernel<256, true><<<gemm_grid, 128, 0, stream>>>(
      S, H1, idg, wl1h, wl1l, wr1h, wr1l, bl1, H2, n);

  // ---- layer 2: H2 -> d_out, no relu ----
  hipMemsetAsync(S, 0, (size_t)n * HID * 4, stream);
  scatter_kernel<256><<<4096, 256, 0, stream>>>(H2, esrc, edst, S, ne);
  sage_gemm_kernel<256, false><<<gemm_grid, 128, 0, stream>>>(
      S, H2, idg, wl2h, wl2l, wr2h, wr2l, bl2, (float*)d_out, n);

  (void)n_in; (void)out_size; (void)ws_size;
}
